// Quantization_66760971649619
// MI455X (gfx1250) — matrix operand order repaired and checked by reading, not compiled, not run on hardware
//
#include <hip/hip_runtime.h>

typedef __attribute__((ext_vector_type(16))) __bf16 v16bf;
typedef __attribute__((ext_vector_type(8)))  __bf16 v8bf;
typedef __attribute__((ext_vector_type(4)))  __bf16 v4bf;
typedef __attribute__((ext_vector_type(8)))  float  v8f;

#define N_ROWS  32768
#define DIMS    512
#define K_CODES 4096

#define MT   64              // rows of x per block (resident in LDS)
#define NT   32              // codes per n-tile
#define DC   64              // d-chunk streamed per step
#define XSTR 520             // 512 + 8 pad (bf16 elems) -> bank spread, 16B aligned
#define CSTR 72              // 64 + 8 pad
#define CBUF (NT * CSTR)     // one cb buffer (hi or lo), elems

// ---------------------------------------------------------------------------
// Kernel 1: split codebook fp32 -> bf16 hi/lo into ws, compute ||c||^2.
// ---------------------------------------------------------------------------
__global__ __launch_bounds__(128)
void vq_prep_kernel(const float* __restrict__ cb,
                    __bf16* __restrict__ cbh, __bf16* __restrict__ cbl,
                    float* __restrict__ cnorm)
{
    __shared__ float red[4];
    const int k = blockIdx.x;
    const int t = threadIdx.x;
    float4 f = ((const float4*)(cb + (size_t)k * DIMS))[t];
    __bf16 h0 = (__bf16)f.x, h1 = (__bf16)f.y, h2 = (__bf16)f.z, h3 = (__bf16)f.w;
    v4bf hv = { h0, h1, h2, h3 };
    v4bf lv = { (__bf16)(f.x - (float)h0), (__bf16)(f.y - (float)h1),
                (__bf16)(f.z - (float)h2), (__bf16)(f.w - (float)h3) };
    *(v4bf*)(cbh + (size_t)k * DIMS + t * 4) = hv;
    *(v4bf*)(cbl + (size_t)k * DIMS + t * 4) = lv;
    float s = f.x * f.x + f.y * f.y + f.z * f.z + f.w * f.w;
    #pragma unroll
    for (int m = 16; m >= 1; m >>= 1) s += __shfl_xor(s, m, 32);
    if ((t & 31) == 0) red[t >> 5] = s;
    __syncthreads();
    if (t == 0) cnorm[k] = red[0] + red[1] + red[2] + red[3];
}

// ---------------------------------------------------------------------------
// Kernel 2: distance GEMM (bf16x3 WMMA) + running argmin over K.
// Block: 256 threads = 8 waves = 4 row-groups x 2 col-groups.
// Register-staged double buffer: global->VGPR prefetch at loop top,
// WMMA math in the shadow of the loads, VGPR->LDS store at loop bottom.
// ---------------------------------------------------------------------------
__global__ __launch_bounds__(256)
void vq_argmin_kernel(const float* __restrict__ x,
                      const __bf16* __restrict__ cbh_g,
                      const __bf16* __restrict__ cbl_g,
                      const float* __restrict__ cnorm,
                      int* __restrict__ ids)
{
    extern __shared__ char smem_raw[];
    __bf16* xh  = (__bf16*)smem_raw;          // [MT][XSTR]
    __bf16* xl  = xh + MT * XSTR;             // [MT][XSTR]
    __bf16* cbh = xl + MT * XSTR;             // [2][NT][CSTR]
    __bf16* cbl = cbh + 2 * CBUF;             // [2][NT][CSTR]

    const int tid  = threadIdx.x;
    const int lane = tid & 31;
    const int w    = tid >> 5;
    const int r    = w >> 1;                  // row group 0..3
    const int c    = w & 1;                   // col group 0..1
    const int sel  = lane >> 4;               // wave half
    const int ln   = lane & 15;
    const int bm   = blockIdx.x * MT;

    // ---- stage resident x tile, split fp32 -> bf16 hi/lo ----
    const float4* xg = (const float4*)(x + (size_t)bm * DIMS);
    for (int idx = tid; idx < MT * (DIMS / 4); idx += 256) {
        int row = idx >> 7;                   // DIMS/4 == 128
        int c4  = idx & 127;
        float4 f = xg[row * 128 + c4];
        __bf16 h0 = (__bf16)f.x, h1 = (__bf16)f.y, h2 = (__bf16)f.z, h3 = (__bf16)f.w;
        v4bf hv = { h0, h1, h2, h3 };
        v4bf lv = { (__bf16)(f.x - (float)h0), (__bf16)(f.y - (float)h1),
                    (__bf16)(f.z - (float)h2), (__bf16)(f.w - (float)h3) };
        int off = row * XSTR + c4 * 4;
        *(v4bf*)(xh + off) = hv;
        *(v4bf*)(xl + off) = lv;
    }

    // cb chunk staging: codes [n0,n0+NT) x d [d0,d0+DC)
    const int crow = tid >> 3;                // 0..31 (code within tile)
    const int ccol = (tid & 7) * 8;           // 0..56 (d within chunk)
    v8bf ph, pl;                              // staged 16B of hi / lo chunk
    auto fetch_chunk = [&](int n0, int d0) {  // global -> VGPR (no LDS yet)
        size_t src = (size_t)(n0 + crow) * DIMS + d0 + ccol;
        ph = *(const v8bf*)(cbh_g + src);
        pl = *(const v8bf*)(cbl_g + src);
    };
    auto stash_chunk = [&](int buf) {         // VGPR -> LDS
        int dst = buf * CBUF + crow * CSTR + ccol;
        *(v8bf*)(cbh + dst) = ph;
        *(v8bf*)(cbl + dst) = pl;
    };

    const int arow = r * 16 + ln;             // x row within block tile
    const int brow = c * 16 + ln;             // code row within n-tile

    float best[8];
    int   bidx[8];
    #pragma unroll
    for (int v = 0; v < 8; ++v) { best[v] = 3.4e38f; bidx[v] = 0; }

    const int NCHUNK = (K_CODES / NT) * (DIMS / DC);   // 1024
    fetch_chunk(0, 0);
    stash_chunk(0);
    __syncthreads();

    v8f acc;
    float cn = 0.f;
    int buf = 0;
    for (int ci = 0; ci < NCHUNK; ++ci) {
        const int n0 = (ci >> 3) * NT;
        const int d0 = (ci & 7) * DC;
        const bool more = (ci + 1 < NCHUNK);

        // issue next chunk's global loads first; math below hides their latency
        if (more) fetch_chunk(((ci + 1) >> 3) * NT, ((ci + 1) & 7) * DC);

        if ((ci & 7) == 0) {
            #pragma unroll
            for (int v = 0; v < 8; ++v) acc[v] = 0.f;
            cn = cnorm[n0 + c * 16 + ln];
        }

        #pragma unroll
        for (int dd = 0; dd < DC; dd += 32) {
            union { v16bf v; v8bf h[2]; } ah, al, bh, bl;
            // A 16x32 bf16: lanes 0-15 K=[0..7,16..23], lanes 16-31 K=[8..15,24..31]
            const int abase = arow * XSTR + d0 + dd + sel * 8;
            ah.h[0] = *(const v8bf*)(xh + abase);
            ah.h[1] = *(const v8bf*)(xh + abase + 16);
            al.h[0] = *(const v8bf*)(xl + abase);
            al.h[1] = *(const v8bf*)(xl + abase + 16);
            // B 32x16 bf16: lanes 0-15 K=[0..7,16..23], lanes 16-31 K=[8..15,24..31], col = ln
            const int bbase = buf * CBUF + brow * CSTR + dd + sel * 8;
            bh.h[0] = *(const v8bf*)(cbh + bbase);
            bh.h[1] = *(const v8bf*)(cbh + bbase + 16);
            bl.h[0] = *(const v8bf*)(cbl + bbase);
            bl.h[1] = *(const v8bf*)(cbl + bbase + 16);
            // dot ~= hi*hi + hi*lo + lo*hi  (lo*lo negligible)
            acc = __builtin_amdgcn_wmma_f32_16x16x32_bf16(false, ah.v, false, bh.v,
                                                          (short)0, acc, false, false);
            acc = __builtin_amdgcn_wmma_f32_16x16x32_bf16(false, ah.v, false, bl.v,
                                                          (short)0, acc, false, false);
            acc = __builtin_amdgcn_wmma_f32_16x16x32_bf16(false, al.v, false, bh.v,
                                                          (short)0, acc, false, false);
        }

        if ((ci & 7) == 7) {
            const int code = n0 + c * 16 + ln;
            #pragma unroll
            for (int v = 0; v < 8; ++v) {
                float dist = cn - 2.0f * acc[v];      // ||c||^2 - 2 x.c
                if (dist < best[v]) { best[v] = dist; bidx[v] = code; }
            }
        }

        // store staged chunk into the other buffer (waits on loads land here)
        if (more) stash_chunk(buf ^ 1);
        __syncthreads();
        buf ^= 1;
    }

    // intra-wave argmin across the 16 columns held by lanes ln=0..15
    #pragma unroll
    for (int v = 0; v < 8; ++v) {
        #pragma unroll
        for (int m = 1; m < 16; m <<= 1) {
            float od = __shfl_xor(best[v], m, 32);
            int   oi = __shfl_xor(bidx[v], m, 32);
            if (od < best[v] || (od == best[v] && oi < bidx[v])) {
                best[v] = od; bidx[v] = oi;
            }
        }
    }

    // combine the two col-groups (c=0,1) per row via LDS (x tile no longer needed)
    float* sbest = (float*)smem_raw;          // [64][2]
    int*   sidx  = (int*)(sbest + MT * 2);    // [64][2]
    if (ln == 0) {
        #pragma unroll
        for (int v = 0; v < 8; ++v) {
            int rloc = r * 16 + v + sel * 8;  // row within block: 0..63
            sbest[rloc * 2 + c] = best[v];
            sidx [rloc * 2 + c] = bidx[v];
        }
    }
    __syncthreads();
    if (tid < MT) {
        float d0 = sbest[tid * 2 + 0], d1 = sbest[tid * 2 + 1];
        int   i0 = sidx [tid * 2 + 0], i1 = sidx [tid * 2 + 1];
        bool take1 = (d1 < d0) || (d1 == d0 && i1 < i0);
        ids[bm + tid] = take1 ? i1 : i0;
    }
}

// ---------------------------------------------------------------------------
// Kernel 3: exact fp32 epilogue: gather emb, emb_out = x + (emb - x),
//           loss = 1.25 * ||x - emb||^2.  One wave per row.
// ---------------------------------------------------------------------------
__global__ __launch_bounds__(256)
void vq_finalize_kernel(const float* __restrict__ x,
                        const float* __restrict__ cb,
                        const int* __restrict__ ids,
                        float* __restrict__ emb_out,
                        float* __restrict__ loss)
{
    const int w    = threadIdx.x >> 5;
    const int lane = threadIdx.x & 31;
    const int row  = blockIdx.x * 8 + w;
    const int id   = ids[row];
    const float4* xr = (const float4*)(x  + (size_t)row * DIMS);
    const float4* er = (const float4*)(cb + (size_t)id  * DIMS);
    float4* outr = (float4*)(emb_out + (size_t)row * DIMS);
    float s = 0.f;
    #pragma unroll
    for (int i = 0; i < 4; ++i) {
        int j = lane + i * 32;
        float4 a = xr[j], e = er[j];
        float dx = a.x - e.x, dy = a.y - e.y, dz = a.z - e.z, dw = a.w - e.w;
        float4 o = { a.x - dx, a.y - dy, a.z - dz, a.w - dw };  // x + (emb - x)
        outr[j] = o;
        s += dx * dx + dy * dy + dz * dz + dw * dw;
    }
    #pragma unroll
    for (int m = 16; m >= 1; m >>= 1) s += __shfl_xor(s, m, 32);
    if (lane == 0) loss[row] = 1.25f * s;
}

// ---------------------------------------------------------------------------
extern "C" void kernel_launch(void* const* d_in, const int* in_sizes, int n_in,
                              void* d_out, int out_size, void* d_ws, size_t ws_size,
                              hipStream_t stream)
{
    (void)in_sizes; (void)n_in; (void)out_size; (void)ws_size;
    const float* x  = (const float*)d_in[0];
    const float* cb = (const float*)d_in[1];

    // d_out layout: emb_out [N,D] f32 | ids [N] i32 | loss [N] f32
    float* emb_out = (float*)d_out;
    int*   ids     = (int*)((float*)d_out + (size_t)N_ROWS * DIMS);
    float* loss    = (float*)d_out + (size_t)N_ROWS * DIMS + N_ROWS;

    // ws layout: cb_hi bf16 [K,D] | cb_lo bf16 [K,D] | cnorm f32 [K]  (~8.02 MB)
    __bf16* cbh   = (__bf16*)d_ws;
    __bf16* cbl   = cbh + (size_t)K_CODES * DIMS;
    float*  cnorm = (float*)(cbl + (size_t)K_CODES * DIMS);

    const size_t lds_bytes = (size_t)(2 * MT * XSTR + 4 * CBUF) * sizeof(__bf16);
    (void)hipFuncSetAttribute((const void*)vq_argmin_kernel,
                              hipFuncAttributeMaxDynamicSharedMemorySize,
                              (int)lds_bytes);

    vq_prep_kernel<<<K_CODES, 128, 0, stream>>>(cb, cbh, cbl, cnorm);
    vq_argmin_kernel<<<N_ROWS / MT, 256, lds_bytes, stream>>>(x, cbh, cbl, cnorm, ids);
    vq_finalize_kernel<<<N_ROWS / 8, 256, 0, stream>>>(x, cb, ids, emb_out, loss);
}
